// Transformer_49452253446800
// MI455X (gfx1250) — compile-verified, matrix operand order checked by reading, not run on hardware
//
#include <hip/hip_runtime.h>
#include <math.h>

// ---------------------------------------------------------------------------
// CompGCN layer, frequency-domain restructure for MI455X (gfx1250, wave32).
//   FX   = x @ DF                  (DFT-as-GEMM, bf16x3-split WMMA)
//   ACC[dst] += conj(FX[src])*Frel[et]*norm   (atomic scatter in freq domain)
//   H    = ACC @ BB + x @ B5 + bias           (irfft+W folded into BB/B5)
//   out  = tanh(batchnorm(H)),  out2 = rel_repr @ w_rel
//
// GEMMs: V_WMMA_F32_16X16X32_BF16 with fp32->bf16 hi/lo splitting.
//   x*y ~= xh*yh + xh*yl + xl*yh   (products exact in fp32 accumulator)
// A operand: global fp32 -> registers -> split -> ds_store (needs VALU).
// B operand: pre-split + pre-transposed bf16 in global (we build it), staged
//            with GLOBAL_LOAD_ASYNC_TO_LDS_B128 (ASYNCcnt) -- pure DMA.
// ---------------------------------------------------------------------------

typedef float v8f  __attribute__((ext_vector_type(8)));
typedef __bf16 v8bf  __attribute__((ext_vector_type(8)));
typedef __bf16 v16bf __attribute__((ext_vector_type(16)));

#define DIM    256
#define NBINS  129        // rfft bins of length-256 real signal
#define FPAD   144        // bins padded to multiple of 16
#define FX_LD  (2*FPAD)   // 288 : [re(144) | im(144)]
#define ACC_LD (4*FPAD)   // 576 : [re_in | im_in | re_out | im_out]
#define TWO_PI_OVER_N 0.0245436926f   // 2*pi/256

// ---------------------------- bf16 split helpers ----------------------------
__device__ __forceinline__ unsigned bf_rne(float f) {   // bf16 bits (RNE)
    unsigned u = __builtin_bit_cast(unsigned, f);
    return (u + 0x7FFFu + ((u >> 16) & 1u)) >> 16;
}
__device__ __forceinline__ unsigned bf_hi(float f, float& rest) {
    unsigned r = bf_rne(f);
    rest = f - __builtin_bit_cast(float, r << 16);
    return r;
}

// ---------------- CDNA5 async global->LDS copy (16B per lane) ---------------
__device__ __forceinline__ void async_b128(unsigned lds_off, unsigned long long gaddr) {
    asm volatile("global_load_async_to_lds_b128 %0, %1, off"
                 :: "v"(lds_off), "v"(gaddr) : "memory");
}
__device__ __forceinline__ void wait_async0() {
    asm volatile("s_wait_asynccnt 0x0" ::: "memory");
}

// ---------------------------- WMMA GEMM ------------------------------------
// C[M x N] = A1[M x K1] @ B1 + A2[M x K2] @ B2 + bias
// B operands given as bf16 hi/lo, transposed [n][k], row stride = K.
// Workgroup tile 128x128, 8 waves -> 32x64 per wave (2x4 C tiles).
#define TM  128
#define TN  128
#define KS  32
#define SK  40            // padded K stride (bf16 elems): 16 lanes x 16B hit all 64 banks

__global__ __launch_bounds__(256) void gemm_bf16x3_wmma(
    const float* __restrict__ A1, int lda1, int K1,
    const unsigned short* __restrict__ B1h, const unsigned short* __restrict__ B1l,
    const float* __restrict__ A2, int lda2, int K2,
    const unsigned short* __restrict__ B2h, const unsigned short* __restrict__ B2l,
    const float* __restrict__ bias,
    float* __restrict__ C, int ldc, int M, int N)
{
    __shared__ __attribute__((aligned(16))) unsigned short sAh[TM * SK]; // A [m][k]
    __shared__ __attribute__((aligned(16))) unsigned short sAl[TM * SK];
    __shared__ __attribute__((aligned(16))) unsigned short sBh[TN * SK]; // B [n][k]
    __shared__ __attribute__((aligned(16))) unsigned short sBl[TN * SK];

    const int tid  = threadIdx.x;
    const int lane = tid & 31;
    const int wv   = tid >> 5;
    const int wm   = (wv & 3) * 32;      // wave M offset (4 groups)
    const int wn   = (wv >> 2) * 64;     // wave N offset (2 groups)
    const int half = lane >> 4;
    const int l16  = lane & 15;
    const int m0   = blockIdx.x * TM;
    const int n0   = blockIdx.y * TN;
    const int c0   = 8 * half;           // 16-bit A/B frag K-chunk base

    const unsigned sBh_base = (unsigned)(uintptr_t)(void*)&sBh[0];
    const unsigned sBl_base = (unsigned)(uintptr_t)(void*)&sBl[0];

    const v8f vzero = {0.f,0.f,0.f,0.f,0.f,0.f,0.f,0.f};
    v8f acc[2][4];
    #pragma unroll
    for (int i = 0; i < 2; ++i)
        #pragma unroll
        for (int j = 0; j < 4; ++j) acc[i][j] = vzero;

    // staging maps
    const int ar  = tid >> 1;            // A row 0..127
    const int ac  = (tid & 1) * 16;      // A col 0 or 16
    const int bq  = tid & 3;             // B 16B chunk within 64B row-slab
    const int br0 = tid >> 2;            // B row 0..63 (rep adds 64)

    for (int s = 0; s < 2; ++s) {
        const float*          A  = s ? A2  : A1;
        const unsigned short* Bh = s ? B2h : B1h;
        const unsigned short* Bl = s ? B2l : B1l;
        const int            lda = s ? lda2 : lda1;
        const int            K   = s ? K2   : K1;
        if (A == nullptr || K <= 0) continue;
        for (int kk = 0; kk < K; kk += KS) {      // K is a multiple of 32
            __syncthreads();
            // ---- B tile (TN x KS) hi/lo: async DMA, already bf16 [n][k] ----
            #pragma unroll
            for (int rep = 0; rep < 2; ++rep) {
                const int row = br0 + rep * 64;
                if (n0 + row < N) {
                    const unsigned long long goff =
                        ((unsigned long long)(n0 + row) * (unsigned)K + (unsigned)kk) * 2ull
                        + (unsigned)(bq * 16);
                    const unsigned loff = (unsigned)(row * SK * 2 + bq * 16);
                    async_b128(sBh_base + loff, (unsigned long long)(uintptr_t)Bh + goff);
                    async_b128(sBl_base + loff, (unsigned long long)(uintptr_t)Bl + goff);
                }
            }
            // ---- A tile (TM x KS): load fp32, split to bf16 hi/lo ----
            {
                float va[16];
                #pragma unroll
                for (int i = 0; i < 16; ++i) va[i] = 0.f;
                if (m0 + ar < M) {
                    const float* ap = A + (size_t)(m0 + ar) * lda + kk + ac;
                    #pragma unroll
                    for (int i = 0; i < 4; ++i) {
                        float4 v = *reinterpret_cast<const float4*>(ap + 4 * i);
                        va[4*i+0] = v.x; va[4*i+1] = v.y; va[4*i+2] = v.z; va[4*i+3] = v.w;
                    }
                }
                unsigned hp[8], lp[8];
                #pragma unroll
                for (int i = 0; i < 8; ++i) {
                    float r0, r1;
                    unsigned h0 = bf_hi(va[2*i], r0), h1 = bf_hi(va[2*i+1], r1);
                    hp[i] = h0 | (h1 << 16);
                    lp[i] = bf_rne(r0) | (bf_rne(r1) << 16);
                }
                uint4* dh = reinterpret_cast<uint4*>(&sAh[ar * SK + ac]);
                uint4* dl = reinterpret_cast<uint4*>(&sAl[ar * SK + ac]);
                dh[0] = make_uint4(hp[0], hp[1], hp[2], hp[3]);
                dh[1] = make_uint4(hp[4], hp[5], hp[6], hp[7]);
                dl[0] = make_uint4(lp[0], lp[1], lp[2], lp[3]);
                dl[1] = make_uint4(lp[4], lp[5], lp[6], lp[7]);
            }
            wait_async0();        // own wave's async copies landed in LDS
            __syncthreads();      // all waves' copies + ds stores visible
            // ---- fragments: lanes 0-15 K{0-7,16-23}, lanes 16-31 K{8-15,24-31} ----
            v16bf ah[2], al[2], bh[4], bl[4];
            #pragma unroll
            for (int im = 0; im < 2; ++im) {
                const unsigned short* ph = &sAh[(wm + im*16 + l16) * SK];
                const unsigned short* pl = &sAl[(wm + im*16 + l16) * SK];
                v8bf h0 = *reinterpret_cast<const v8bf*>(ph + c0);
                v8bf h1 = *reinterpret_cast<const v8bf*>(ph + c0 + 16);
                v8bf l0 = *reinterpret_cast<const v8bf*>(pl + c0);
                v8bf l1 = *reinterpret_cast<const v8bf*>(pl + c0 + 16);
                ah[im] = __builtin_shufflevector(h0, h1, 0,1,2,3,4,5,6,7,8,9,10,11,12,13,14,15);
                al[im] = __builtin_shufflevector(l0, l1, 0,1,2,3,4,5,6,7,8,9,10,11,12,13,14,15);
            }
            #pragma unroll
            for (int jn = 0; jn < 4; ++jn) {
                const unsigned short* ph = &sBh[(wn + jn*16 + l16) * SK];
                const unsigned short* pl = &sBl[(wn + jn*16 + l16) * SK];
                v8bf h0 = *reinterpret_cast<const v8bf*>(ph + c0);
                v8bf h1 = *reinterpret_cast<const v8bf*>(ph + c0 + 16);
                v8bf l0 = *reinterpret_cast<const v8bf*>(pl + c0);
                v8bf l1 = *reinterpret_cast<const v8bf*>(pl + c0 + 16);
                bh[jn] = __builtin_shufflevector(h0, h1, 0,1,2,3,4,5,6,7,8,9,10,11,12,13,14,15);
                bl[jn] = __builtin_shufflevector(l0, l1, 0,1,2,3,4,5,6,7,8,9,10,11,12,13,14,15);
            }
            // ---- bf16x3 split products, fp32 accumulate ----
            #pragma unroll
            for (int im = 0; im < 2; ++im) {
                #pragma unroll
                for (int jn = 0; jn < 4; ++jn) {
                    v8f a = acc[im][jn];
                    a = __builtin_amdgcn_wmma_f32_16x16x32_bf16(false, ah[im], false, bh[jn], (short)0, a, false, false);
                    a = __builtin_amdgcn_wmma_f32_16x16x32_bf16(false, ah[im], false, bl[jn], (short)0, a, false, false);
                    a = __builtin_amdgcn_wmma_f32_16x16x32_bf16(false, al[im], false, bh[jn], (short)0, a, false, false);
                    acc[im][jn] = a;
                }
            }
        }
    }

    // ---- epilogue: C layout (VGPR i: m=i+8*half, n=l16) ----
    #pragma unroll
    for (int jn = 0; jn < 4; ++jn) {
        const int nn = n0 + wn + jn * 16 + l16;
        if (nn >= N) continue;
        const float bv = bias ? bias[nn] : 0.f;
        #pragma unroll
        for (int im = 0; im < 2; ++im) {
            #pragma unroll
            for (int i = 0; i < 8; ++i) {
                const int mm = m0 + wm + im * 16 + i + 8 * half;
                if (mm < M) C[(size_t)mm * ldc + nn] = acc[im][jn][i] + bv;
            }
        }
    }
}

// ------------------------- precompute kernels -------------------------------
// DF[k][j] fp32 (for Frel), plus transposed bf16 hi/lo DFt[j][k] for the GEMM.
__global__ void build_df(float* __restrict__ DF,
                         unsigned short* __restrict__ DFt_h,
                         unsigned short* __restrict__ DFt_l)
{
    int idx = blockIdx.x * blockDim.x + threadIdx.x;
    if (idx >= DIM * FX_LD) return;
    int k = idx / FX_LD, j = idx % FX_LD;
    int grp = j / FPAD, f = j % FPAD;
    float v = 0.f;
    if (f < NBINS) {
        int ph = (k * f) & 255;                 // exact mod-N phase
        float ang = (float)ph * TWO_PI_OVER_N;
        v = (grp == 0) ? cosf(ang) : -sinf(ang);
    }
    DF[idx] = v;
    float rest; unsigned h = bf_hi(v, rest);
    DFt_h[(size_t)j * DIM + k] = (unsigned short)h;
    DFt_l[(size_t)j * DIM + k] = (unsigned short)bf_rne(rest);
}

// BBt[n][576] bf16 hi/lo: row g*144+f -> {cos,-sin}*{in_w,out_w} * (w_f/256)/3
__global__ __launch_bounds__(256) void build_bb(const float* __restrict__ in_w,
                                                const float* __restrict__ out_w,
                                                unsigned short* __restrict__ BBt_h,
                                                unsigned short* __restrict__ BBt_l)
{
    __shared__ float cs[DIM], sn[DIM];
    int t = threadIdx.x;
    float ang = (float)t * TWO_PI_OVER_N;
    cs[t] = cosf(ang); sn[t] = sinf(ang);
    __syncthreads();
    int idx = blockIdx.x * blockDim.x + t;      // one block per BB row
    if (idx >= ACC_LD * DIM) return;
    int row = idx / DIM, n = idx % DIM;
    int g = row / FPAD, f = row % FPAD;
    float outv = 0.f;
    if (f < NBINS) {
        const float* W = (g < 2) ? in_w : out_w;
        float sc = ((f == 0 || f == 128) ? 1.f : 2.f) * (1.f / (256.f * 3.f));
        bool im = (g & 1);
        float sum = 0.f;
        for (int k = 0; k < DIM; ++k) {
            int ph = (k * f) & 255;
            float c = im ? -sn[ph] : cs[ph];
            sum += c * W[k * DIM + n];
        }
        outv = sum * sc;
    }
    float rest; unsigned h = bf_hi(outv, rest);
    BBt_h[(size_t)n * ACC_LD + row] = (unsigned short)h;
    BBt_l[(size_t)n * ACC_LD + row] = (unsigned short)bf_rne(rest);
}

// B5t[n][j] bf16 hi/lo: B5[j][n] = sum_k loop_rel[(j+k)%256]*loop_w[k][n]/3
__global__ __launch_bounds__(256) void build_b5(const float* __restrict__ loop_rel,
                                                const float* __restrict__ loop_w,
                                                unsigned short* __restrict__ B5t_h,
                                                unsigned short* __restrict__ B5t_l)
{
    __shared__ float lr[DIM];
    int t = threadIdx.x;
    lr[t] = loop_rel[t];
    __syncthreads();
    int idx = blockIdx.x * blockDim.x + t;
    int j = idx / DIM, n = idx % DIM;
    float sum = 0.f;
    for (int k = 0; k < DIM; ++k) sum += lr[(j + k) & 255] * loop_w[k * DIM + n];
    sum *= (1.f / 3.f);
    float rest; unsigned h = bf_hi(sum, rest);
    B5t_h[(size_t)n * DIM + j] = (unsigned short)h;
    B5t_l[(size_t)n * DIM + j] = (unsigned short)bf_rne(rest);
}

// Frel[r][j] = rel_repr[r] @ DF  (500 x 288, tiny, fp32)
__global__ void frel_kernel(const float* __restrict__ rel, const float* __restrict__ DF,
                            float* __restrict__ Frel, int nrels)
{
    int idx = blockIdx.x * blockDim.x + threadIdx.x;
    if (idx >= nrels * FX_LD) return;
    int r = idx / FX_LD, j = idx % FX_LD;
    float sum = 0.f;
    for (int k = 0; k < DIM; ++k) sum += rel[r * DIM + k] * DF[k * FX_LD + j];
    Frel[idx] = sum;
}

// ---------------------- per-edge frequency scatter --------------------------
// one wave per edge; lane f handles bins f, f+32, ... (<129)
__global__ __launch_bounds__(256) void edge_scatter(
    const int* __restrict__ src, const int* __restrict__ dst,
    const int* __restrict__ et,  const float* __restrict__ norm,
    const float* __restrict__ FX, const float* __restrict__ Frel,
    float* __restrict__ ACC, int n_edges, int half)
{
    int gwave = (blockIdx.x * blockDim.x + threadIdx.x) >> 5;
    int lane  = threadIdx.x & 31;
    int nw    = (gridDim.x * blockDim.x) >> 5;
    for (int e = gwave; e < n_edges; e += nw) {
        int s = src[e], d = dst[e], r = et[e];
        float w = norm[e];
        const float* fx = FX   + (size_t)s * FX_LD;
        const float* fr = Frel + (size_t)r * FX_LD;
        float* acc = ACC + (size_t)d * ACC_LD + ((e < half) ? 0 : FX_LD);
        for (int f = lane; f < NBINS; f += 32) {
            float xr = fx[f],      xi = fx[FPAD + f];
            float rr = fr[f],      ri = fr[FPAD + f];
            // conj(Fx) * Frel
            unsafeAtomicAdd(acc + f,        (xr * rr + xi * ri) * w);
            unsafeAtomicAdd(acc + FPAD + f, (xr * ri - xi * rr) * w);
        }
    }
}

// --------------------------- batchnorm + tanh -------------------------------
__global__ __launch_bounds__(256) void bn_stats(const float* __restrict__ H,
                                                float* __restrict__ stats, int M)
{
    int c = threadIdx.x;                       // one column per thread
    float s = 0.f, sq = 0.f;
    for (int r = blockIdx.x; r < M; r += gridDim.x) {
        float v = H[(size_t)r * DIM + c];
        s += v; sq += v * v;
    }
    unsafeAtomicAdd(&stats[c], s);
    unsafeAtomicAdd(&stats[DIM + c], sq);
}

__global__ void bn_final(float* __restrict__ stats, const float* __restrict__ gamma,
                         const float* __restrict__ beta, int M)
{
    int c = threadIdx.x;
    float mean = stats[c] / (float)M;
    float var  = stats[DIM + c] / (float)M - mean * mean;
    float sc   = gamma[c] * rsqrtf(var + 1e-5f);
    stats[2 * DIM + c] = sc;
    stats[3 * DIM + c] = beta[c] - mean * sc;
}

__global__ __launch_bounds__(256) void act_kernel(const float* __restrict__ H,
                                                  const float* __restrict__ sc,
                                                  const float* __restrict__ sh,
                                                  float* __restrict__ out, size_t n)
{
    size_t i = (size_t)blockIdx.x * blockDim.x + threadIdx.x;
    size_t stride = (size_t)gridDim.x * blockDim.x;
    for (; i < n; i += stride) {
        int c = (int)(i & (DIM - 1));
        out[i] = tanhf(H[i] * sc[c] + sh[c]);
    }
}

// rel_out = rel_repr @ w_rel  (500 x 256, tiny)
__global__ void relout_kernel(const float* __restrict__ rel, const float* __restrict__ w_rel,
                              float* __restrict__ out, int nrels)
{
    int idx = blockIdx.x * blockDim.x + threadIdx.x;
    if (idx >= nrels * DIM) return;
    int r = idx / DIM, n = idx % DIM;
    float sum = 0.f;
    for (int k = 0; k < DIM; ++k) sum += rel[r * DIM + k] * w_rel[k * DIM + n];
    out[idx] = sum;
}

// ------------------------------- launcher -----------------------------------
extern "C" void kernel_launch(void* const* d_in, const int* in_sizes, int n_in,
                              void* d_out, int out_size, void* d_ws, size_t ws_size,
                              hipStream_t stream)
{
    const float* x        = (const float*)d_in[0];
    const float* rel_repr = (const float*)d_in[1];
    const float* in_w     = (const float*)d_in[2];
    const float* out_w    = (const float*)d_in[3];
    const float* loop_w   = (const float*)d_in[4];
    const float* w_rel    = (const float*)d_in[5];
    const float* loop_rel = (const float*)d_in[6];
    const float* bias     = (const float*)d_in[7];
    const float* gamma    = (const float*)d_in[8];
    const float* beta     = (const float*)d_in[9];
    const int*   src      = (const int*)d_in[10];
    const int*   dst      = (const int*)d_in[11];
    const int*   et       = (const int*)d_in[12];
    const float* norm     = (const float*)d_in[13];

    const int n_nodes = in_sizes[0] / DIM;     // 100000
    const int n_rels  = in_sizes[1] / DIM;     // 500
    const int n_edges = in_sizes[10];          // 320000
    const int half    = n_edges / 2;

    // ---- workspace layout (float-granular offsets) ----
    float* ws = (float*)d_ws;
    size_t off = 0;
    float* DF    = ws + off; off += (size_t)DIM * FX_LD;            //  73,728 f
    unsigned short* DFt_h = (unsigned short*)(ws + off); off += (size_t)DIM * FX_LD / 2;
    unsigned short* DFt_l = (unsigned short*)(ws + off); off += (size_t)DIM * FX_LD / 2;
    unsigned short* BBt_h = (unsigned short*)(ws + off); off += (size_t)ACC_LD * DIM / 2;
    unsigned short* BBt_l = (unsigned short*)(ws + off); off += (size_t)ACC_LD * DIM / 2;
    unsigned short* B5t_h = (unsigned short*)(ws + off); off += (size_t)DIM * DIM / 2;
    unsigned short* B5t_l = (unsigned short*)(ws + off); off += (size_t)DIM * DIM / 2;
    float* Frel  = ws + off; off += (size_t)n_rels * FX_LD;         // 144,000 f
    float* stats = ws + off; off += 4 * DIM;
    float* ACC   = ws + off; off += (size_t)n_nodes * ACC_LD;       // 57.6 M f
    float* FXH   = ws + off; off += (size_t)n_nodes * FX_LD;        // 28.8 M f
    if (ws_size < off * sizeof(float)) return;                      // workspace too small
    float* H = FXH;   // FX is dead after edge_scatter; H (100000x256) aliases it

    hipMemsetAsync(ACC,   0, (size_t)n_nodes * ACC_LD * sizeof(float), stream);
    hipMemsetAsync(stats, 0, 4 * DIM * sizeof(float), stream);

    build_df <<<(DIM * FX_LD + 255) / 256, 256, 0, stream>>>(DF, DFt_h, DFt_l);
    build_bb <<<ACC_LD, 256, 0, stream>>>(in_w, out_w, BBt_h, BBt_l);
    build_b5 <<<DIM,    256, 0, stream>>>(loop_rel, loop_w, B5t_h, B5t_l);
    frel_kernel<<<(n_rels * FX_LD + 255) / 256, 256, 0, stream>>>(rel_repr, DF, Frel, n_rels);

    // FX = x @ DF   (100000 x 256 @ 256 x 288)
    dim3 g1((n_nodes + TM - 1) / TM, (FX_LD + TN - 1) / TN);
    gemm_bf16x3_wmma<<<g1, 256, 0, stream>>>(x, DIM, DIM, DFt_h, DFt_l,
                                             nullptr, 0, 0, nullptr, nullptr,
                                             nullptr, FXH, FX_LD, n_nodes, FX_LD);

    edge_scatter<<<2048, 256, 0, stream>>>(src, dst, et, norm, FXH, Frel, ACC, n_edges, half);

    // H = ACC @ BB + x @ B5 + bias   (100000 x 256)
    dim3 g2((n_nodes + TM - 1) / TM, (DIM + TN - 1) / TN);
    gemm_bf16x3_wmma<<<g2, 256, 0, stream>>>(ACC, ACC_LD, ACC_LD, BBt_h, BBt_l,
                                             x, DIM, DIM, B5t_h, B5t_l,
                                             bias, H, DIM, n_nodes, DIM);

    bn_stats <<<768, 256, 0, stream>>>(H, stats, n_nodes);
    bn_final <<<1,   256, 0, stream>>>(stats, gamma, beta, n_nodes);
    act_kernel<<<4096, 256, 0, stream>>>(H, stats + 2 * DIM, stats + 3 * DIM,
                                         (float*)d_out, (size_t)n_nodes * DIM);
    relout_kernel<<<(n_rels * DIM + 255) / 256, 256, 0, stream>>>(
        rel_repr, w_rel, (float*)d_out + (size_t)n_nodes * DIM, n_rels);
}
